// DGCNN_type1_7438883356710
// MI455X (gfx1250) — matrix operand order repaired and checked by reading, not compiled, not run on hardware
//
#include <hip/hip_runtime.h>

// ---------------------------------------------------------------------------
// DGCNN forward for MI455X (gfx1250): bf16 WMMA everywhere GEMM-shaped,
// LDS-fused EdgeConv (gather -> MLP -> K-max), fused lin1 + global max pool.
// Round 3: bias folded into WMMA accumulator; leaky = fmax(x, 0.01x);
// native bf16 cvt; async global->LDS staging (ASYNCcnt path).
// ---------------------------------------------------------------------------

#define NB   128   // graphs
#define NN   512   // nodes per graph
#define NKNN 16    // neighbors
#define NEGS 0.01f

typedef __attribute__((ext_vector_type(16))) __bf16 v16bf;
typedef __attribute__((ext_vector_type(8)))  float  v8f;

// identical to where(x>0, x, 0.01x): mul + max, dual-issuable
__device__ __forceinline__ float leakyf(float x) { return fmaxf(x, NEGS * x); }

__device__ __forceinline__ unsigned short f2bf(float f) {
  union { __bf16 h; unsigned short u; } cv;
  cv.h = (__bf16)f;                       // native v_cvt (RNE)
  return cv.u;
}
__device__ __forceinline__ float bf2f(unsigned short h) {
  return __uint_as_float(((unsigned int)h) << 16);
}

union FragU { unsigned int w[8]; v16bf v; };

// A-fragment 16x32 bf16 (ISA 7.12.2): lane -> M=lane&15; vgpr v, lane-half -> K.
// K pairs are contiguous -> one b32 load per vgpr.
__device__ __forceinline__ v16bf load_a(const unsigned short* base, int lane, int stride) {
  const unsigned short* p = base + (lane & 15) * stride;
  int kh = (lane >> 4) << 3;                // 0 | 8
  FragU f;
#pragma unroll
  for (int r = 0; r < 8; ++r) {
    int kb = ((r & 4) << 2) + kh + ((r & 3) << 1);   // (r>=4?16:0)+kh+2*(r&3)
    f.w[r] = *(const unsigned int*)(p + kb);
  }
  return f.v;
}

// B-fragment 32x16 bf16 from a TRANSPOSED (col-major) matrix bT[n][k]:
// lane -> N=lane&15; vgpr v, lane-half -> K (same as A); K pairs contiguous.
__device__ __forceinline__ v16bf load_b(const unsigned short* baseT, int lane, int stride) {
  const unsigned short* p = baseT + (lane & 15) * stride + ((lane >> 4) << 3);
  FragU f;
#pragma unroll
  for (int r = 0; r < 8; ++r) f.w[r] = *(const unsigned int*)(p + ((r & 4) << 2) + ((r & 3) << 1));
  return f.v;
}

__device__ __forceinline__ v8f wmma_bf(v16bf a, v16bf b, v8f c) {
  return __builtin_amdgcn_wmma_f32_16x16x32_bf16(false, a, false, b, (short)0, c, false, false);
}

// bias[n] is per-column constant -> seed all 8 accumulator rows with it
__device__ __forceinline__ v8f acc_bias(float bias) {
  v8f a;
#pragma unroll
  for (int v = 0; v < 8; ++v) a[v] = bias;
  return a;
}

__device__ __forceinline__ void atomicMaxF(float* addr, float val) {
  unsigned int* ua = (unsigned int*)addr;
  unsigned int cur = *ua;
  while (__uint_as_float(cur) < val) {
    unsigned int prev = atomicCAS(ua, cur, __float_as_uint(val));
    if (prev == cur) break;
    cur = prev;
  }
}

// --- CDNA5 async global->LDS copy (ASYNCcnt-tracked, bypasses VGPRs) -------
// Per ISA 10.2: flat address low 32 bits are the LDS byte offset, so the
// truncated generic pointer of a __shared__ object is a valid VDST operand.
__device__ __forceinline__ void async_copy_b128(void* lds_dst, const void* gsrc) {
  unsigned int loff = (unsigned int)(size_t)lds_dst;
  unsigned long long ga = (unsigned long long)(size_t)gsrc;
  asm volatile("global_load_async_to_lds_b128 %0, %1, off"
               :: "v"(loff), "v"(ga) : "memory");
}
__device__ __forceinline__ void async_wait0() {
  asm volatile("s_wait_asynccnt 0x0" ::: "memory");
}

// --------------------------- small prep kernels ----------------------------

// weights: f32 [fin,fout] row-major -> bf16 transposed+K-padded [fout][kpad]
__global__ void k_convw(const float* w, unsigned short* wt, int fin, int fout, int kpad) {
  int i = blockIdx.x * blockDim.x + threadIdx.x;
  if (i < fout * kpad) {
    int o = i / kpad, k = i % kpad;
    wt[i] = (k < fin) ? f2bf(w[k * fout + o]) : (unsigned short)0;
  }
}

__global__ void k_build_xx(const float* x, const float* pos, const float* tq, float* xx) {
  int r = blockIdx.x * blockDim.x + threadIdx.x;
  if (r < NB * NN) {
    xx[r * 5 + 0] = tq[r];
    xx[r * 5 + 1] = x[r];
    xx[r * 5 + 2] = pos[r * 3 + 0];
    xx[r * 5 + 3] = pos[r * 3 + 1];
    xx[r * 5 + 4] = pos[r * 3 + 2];
  }
}

// ------------------------------- kNN (d=5) ---------------------------------
__global__ __launch_bounds__(512) void k_knn1(const float* xx, int* idx1) {
  __shared__ float F[NN * 5];               // 10 KB (= 640 x b128)
  int g = blockIdx.x, t = threadIdx.x;
  const char* src = (const char*)(xx + (size_t)g * NN * 5);
  for (int i = t; i < 640; i += 512)
    async_copy_b128((char*)F + i * 16, src + (size_t)i * 16);
  async_wait0();
  __syncthreads();
  float fi0 = F[t*5], fi1 = F[t*5+1], fi2 = F[t*5+2], fi3 = F[t*5+3], fi4 = F[t*5+4];
  float bd[NKNN]; int bi[NKNN];
#pragma unroll
  for (int k = 0; k < NKNN; ++k) { bd[k] = __builtin_inff(); bi[k] = -1; }
  for (int j = 0; j < NN; ++j) {
    const float* fj = &F[j * 5];
    float d0 = fj[0]-fi0, d1 = fj[1]-fi1, d2 = fj[2]-fi2, d3 = fj[3]-fi3, d4 = fj[4]-fi4;
    float dd = d0*d0 + d1*d1 + d2*d2 + d3*d3 + d4*d4;
    if (dd < bd[NKNN-1]) {
      int p = NKNN - 1;
      while (p > 0 && bd[p-1] > dd) { bd[p] = bd[p-1]; bi[p] = bi[p-1]; --p; }
      bd[p] = dd; bi[p] = j;
    }
  }
  int* out = idx1 + ((size_t)g * NN + t) * NKNN;
#pragma unroll
  for (int k = 0; k < NKNN; ++k) out[k] = bi[k];
}

// -------------------------- EdgeConv1 (fused) ------------------------------
// Per WG: 16 nodes -> 256 edges. feat 10 (pad 32) -> 64 -> 64, max over K=16.
// Edge rows arranged so m-tile == local node: K-max = 7 v_max + shfl_xor(16).
__global__ __launch_bounds__(256) void k_edgeconv1(const float* xx, const int* idx1,
    const unsigned short* WaT, const float* ba, const unsigned short* WbT, const float* bb,
    unsigned short* x1) {
  __shared__ float F[NN * 5];               // 10 KB
  __shared__ unsigned short A[256 * 32];    // 16 KB
  __shared__ unsigned short H[256 * 64];    // 32 KB
  int g = blockIdx.x >> 5, tile = blockIdx.x & 31;
  int t = threadIdx.x, lane = t & 31, wave = t >> 5;
  const char* src = (const char*)(xx + (size_t)g * NN * 5);
  for (int i = t; i < 640; i += 256)
    async_copy_b128((char*)F + i * 16, src + (size_t)i * 16);
  async_wait0();
  __syncthreads();
  {
    int e = t, ln = e >> 4, kk = e & 15;
    int i = tile * 16 + ln;
    int j = idx1[((size_t)g * NN + i) * NKNN + kk];
    unsigned short* ar = &A[e * 32];
#pragma unroll
    for (int c = 0; c < 5; ++c) {
      float xi = F[i * 5 + c], xj = F[j * 5 + c];
      ar[c] = f2bf(xi);
      ar[5 + c] = f2bf(xj - xi);
    }
#pragma unroll
    for (int c = 10; c < 32; ++c) ar[c] = 0;
  }
  __syncthreads();
  for (int tt = wave; tt < 64; tt += 8) {   // layer a: 16m x 4n, K=32
    int mt = tt >> 2, nt = tt & 3;
    int n = nt * 16 + (lane & 15);
    v8f acc = acc_bias(ba[n]);
    v16bf a = load_a(&A[mt * 16 * 32], lane, 32);
    v16bf b = load_b(WaT + nt * 16 * 32, lane, 32);
    acc = wmma_bf(a, b, acc);
    int mb = mt * 16 + ((lane >> 4) << 3);
#pragma unroll
    for (int v = 0; v < 8; ++v) H[(mb + v) * 64 + n] = f2bf(leakyf(acc[v]));
  }
  __syncthreads();
  for (int tt = wave; tt < 64; tt += 8) {   // layer b: 16m x 4n, K=64 + K-max
    int mt = tt >> 2, nt = tt & 3;
    int n = nt * 16 + (lane & 15);
    v8f acc = acc_bias(bb[n]);
#pragma unroll
    for (int kk = 0; kk < 2; ++kk) {
      v16bf a = load_a(&H[mt * 16 * 64 + kk * 32], lane, 64);
      v16bf b = load_b(WbT + nt * 16 * 64 + kk * 32, lane, 64);
      acc = wmma_bf(a, b, acc);
    }
    float mx = -__builtin_inff();
#pragma unroll
    for (int v = 0; v < 8; ++v) mx = fmaxf(mx, leakyf(acc[v]));
    mx = fmaxf(mx, __shfl_xor(mx, 16));
    if (lane < 16) {
      int node = tile * 16 + mt;
      x1[((size_t)g * NN + node) * 64 + n] = f2bf(mx);
    }
  }
}

// ---------------------------- kNN (d=64) -----------------------------------
__global__ void k_sqnorm(const unsigned short* x1, float* sq) {
  int r = blockIdx.x * blockDim.x + threadIdx.x;
  if (r < NB * NN) {
    const unsigned short* p = x1 + (size_t)r * 64;
    float s = 0.f;
    for (int c = 0; c < 64; ++c) { float v = bf2f(p[c]); s += v * v; }
    sq[r] = s;
  }
}

// Gram via WMMA (B operand of X@X^T == row-major X), d2 strip in LDS, top-16.
__global__ __launch_bounds__(256) void k_knn2(const unsigned short* x1, const float* sq, int* idx2) {
  __shared__ float D[32 * NN];              // 64 KB
  int g = blockIdx.x >> 4, chunk = blockIdx.x & 15;
  int lane = threadIdx.x & 31, wave = threadIdx.x >> 5;
  const unsigned short* Xg = x1 + (size_t)g * NN * 64;
  const float* sqg = sq + g * NN;
  int row0 = chunk * 32;
  for (int tt = wave; tt < 64; tt += 8) {   // 2m x 32n tiles, K=64
    int mt = tt >> 5, nt = tt & 31;
    v8f acc = {};
#pragma unroll
    for (int kk = 0; kk < 2; ++kk) {
      v16bf a = load_a(Xg + (size_t)(row0 + mt * 16) * 64 + kk * 32, lane, 64);
      v16bf b = load_b(Xg + (size_t)(nt * 16) * 64 + kk * 32, lane, 64);
      acc = wmma_bf(a, b, acc);
    }
    int n = nt * 16 + (lane & 15);
    int mb = mt * 16 + ((lane >> 4) << 3);
    float sqn = sqg[n];
#pragma unroll
    for (int v = 0; v < 8; ++v) {
      int i = mb + v;
      D[i * NN + n] = sqg[row0 + i] + sqn - 2.0f * acc[v];
    }
  }
  __syncthreads();
  if (threadIdx.x < 32) {
    const float* row = &D[threadIdx.x * NN];
    float bd[NKNN]; int bi[NKNN];
#pragma unroll
    for (int k = 0; k < NKNN; ++k) { bd[k] = __builtin_inff(); bi[k] = -1; }
    for (int j = 0; j < NN; ++j) {
      float dd = row[j];
      if (dd < bd[NKNN-1]) {
        int p = NKNN - 1;
        while (p > 0 && bd[p-1] > dd) { bd[p] = bd[p-1]; bi[p] = bi[p-1]; --p; }
        bd[p] = dd; bi[p] = j;
      }
    }
    int* out = idx2 + ((size_t)g * NN + row0 + threadIdx.x) * NKNN;
#pragma unroll
    for (int k = 0; k < NKNN; ++k) out[k] = bi[k];
  }
}

// -------------------------- EdgeConv2 (fused) ------------------------------
// Per WG: 8 nodes -> 128 edges. feat 128 -> 128 -> 64, max over K=16.
__global__ __launch_bounds__(256) void k_edgeconv2(const unsigned short* x1, const int* idx2,
    const unsigned short* WaT, const float* ba, const unsigned short* WbT, const float* bb,
    unsigned short* x2) {
  __shared__ unsigned short A[128 * 128];   // 32 KB
  __shared__ unsigned short H[128 * 128];   // 32 KB
  int g = blockIdx.x >> 6, tile = blockIdx.x & 63;
  int t = threadIdx.x, lane = t & 31, wave = t >> 5;
  const unsigned short* Xg = x1 + (size_t)g * NN * 64;
  {
    int e = t >> 1, half = t & 1;           // 2 threads per edge
    int ln = e >> 4, kk = e & 15;
    int i = tile * 8 + ln;
    int j = idx2[((size_t)g * NN + i) * NKNN + kk];
    const unsigned short* xi = Xg + (size_t)i * 64;
    const unsigned short* xj = Xg + (size_t)j * 64;
    unsigned short* ar = &A[e * 128];
    for (int c = half * 32; c < half * 32 + 32; ++c) {
      ar[c] = xi[c];
      ar[64 + c] = f2bf(bf2f(xj[c]) - bf2f(xi[c]));
    }
  }
  __syncthreads();
  for (int tt = wave; tt < 64; tt += 8) {   // layer a: 8m x 8n, K=128
    int mt = tt >> 3, nt = tt & 7;
    int n = nt * 16 + (lane & 15);
    v8f acc = acc_bias(ba[n]);
#pragma unroll
    for (int kk = 0; kk < 4; ++kk) {
      v16bf a = load_a(&A[mt * 16 * 128 + kk * 32], lane, 128);
      v16bf b = load_b(WaT + nt * 16 * 128 + kk * 32, lane, 128);
      acc = wmma_bf(a, b, acc);
    }
    int mb = mt * 16 + ((lane >> 4) << 3);
#pragma unroll
    for (int v = 0; v < 8; ++v) H[(mb + v) * 128 + n] = f2bf(leakyf(acc[v]));
  }
  __syncthreads();
  for (int tt = wave; tt < 32; tt += 8) {   // layer b: 8m x 4n, K=128 + K-max
    int mt = tt >> 2, nt = tt & 3;
    int n = nt * 16 + (lane & 15);
    v8f acc = acc_bias(bb[n]);
#pragma unroll
    for (int kk = 0; kk < 4; ++kk) {
      v16bf a = load_a(&H[mt * 16 * 128 + kk * 32], lane, 128);
      v16bf b = load_b(WbT + nt * 16 * 128 + kk * 32, lane, 128);
      acc = wmma_bf(a, b, acc);
    }
    float mx = -__builtin_inff();
#pragma unroll
    for (int v = 0; v < 8; ++v) mx = fmaxf(mx, leakyf(acc[v]));
    mx = fmaxf(mx, __shfl_xor(mx, 16));
    if (lane < 16) {
      int node = tile * 8 + mt;
      x2[((size_t)g * NN + node) * 64 + n] = f2bf(mx);
    }
  }
}

// ------------------------------ comb + lin1 --------------------------------
__global__ void k_comb(const float* xx, const unsigned short* x1, const unsigned short* x2,
                       unsigned short* comb) {
  int idx = blockIdx.x * blockDim.x + threadIdx.x;
  if (idx < NB * NN * 160) {
    int r = idx / 160, c = idx % 160;
    unsigned short v;
    if (c < 5)        v = f2bf(xx[r * 5 + c]);
    else if (c < 69)  v = x1[(size_t)r * 64 + (c - 5)];
    else if (c < 133) v = x2[(size_t)r * 64 + (c - 69)];
    else              v = 0;
    comb[idx] = v;
  }
}

__global__ void k_initg(float* gbuf) {
  int i = blockIdx.x * blockDim.x + threadIdx.x;
  if (i < NB * 256) gbuf[i] = -__builtin_inff();
}

// 32 rows/WG: leaky(comb@wl1+bl1) in LDS, @wl2+bl2, row-max -> atomicMax pool.
__global__ __launch_bounds__(256) void k_lin1(const unsigned short* comb,
    const unsigned short* W1T, const float* b1, const unsigned short* W2T, const float* b2,
    float* gbuf) {
  __shared__ unsigned short A1[32 * 160];   // 10 KB (= 640 x b128)
  __shared__ unsigned short H[32 * 512];    // 32 KB
  int row0 = blockIdx.x * 32;
  int graph = row0 >> 9;
  int t = threadIdx.x, lane = t & 31, wave = t >> 5;
  const char* src = (const char*)(comb + (size_t)row0 * 160);
  for (int i = t; i < 640; i += 256)
    async_copy_b128((char*)A1 + i * 16, src + (size_t)i * 16);
  async_wait0();
  __syncthreads();
  for (int tt = wave; tt < 64; tt += 8) {   // phase 1: 2m x 32n, K=160
    int mt = tt >> 5, nt = tt & 31;
    int n = nt * 16 + (lane & 15);
    v8f acc = acc_bias(b1[n]);
#pragma unroll
    for (int kk = 0; kk < 5; ++kk) {
      v16bf a = load_a(&A1[mt * 16 * 160 + kk * 32], lane, 160);
      v16bf b = load_b(W1T + nt * 16 * 160 + kk * 32, lane, 160);
      acc = wmma_bf(a, b, acc);
    }
    int mb = mt * 16 + ((lane >> 4) << 3);
#pragma unroll
    for (int v = 0; v < 8; ++v) H[(mb + v) * 512 + n] = f2bf(leakyf(acc[v]));
  }
  __syncthreads();
  for (int tt = wave; tt < 32; tt += 8) {   // phase 2: 2m x 16n, K=512 + pool
    int mt = tt >> 4, nt = tt & 15;
    int n = nt * 16 + (lane & 15);
    v8f acc = acc_bias(b2[n]);
    for (int kk = 0; kk < 16; ++kk) {
      v16bf a = load_a(&H[mt * 16 * 512 + kk * 32], lane, 512);
      v16bf b = load_b(W2T + nt * 16 * 512 + kk * 32, lane, 512);
      acc = wmma_bf(a, b, acc);
    }
    float mx = -__builtin_inff();
#pragma unroll
    for (int v = 0; v < 8; ++v) mx = fmaxf(mx, acc[v]);
    mx = fmaxf(mx, __shfl_xor(mx, 16));
    if (lane < 16) atomicMaxF(&gbuf[graph * 256 + n], mx);
  }
}

// ---------------------------------- head -----------------------------------
__global__ __launch_bounds__(128) void k_head(const float* gbuf,
    const float* wm1, const float* bm1, const float* wm2, const float* bm2, float* out) {
  __shared__ float grow[256];
  __shared__ float hid[128];
  int g = blockIdx.x, t = threadIdx.x;
  for (int i = t; i < 256; i += 128) grow[i] = leakyf(gbuf[g * 256 + i]);
  __syncthreads();
  float s = bm1[t];
  for (int k = 0; k < 256; ++k) s += grow[k] * wm1[k * 128 + t];
  hid[t] = leakyf(s);
  __syncthreads();
  if (t < 3) {
    float o = bm2[t];
    for (int k = 0; k < 128; ++k) o += hid[k] * wm2[k * 3 + t];
    out[g * 3 + t] = o;
  }
}

// ------------------------------- launcher ----------------------------------
extern "C" void kernel_launch(void* const* d_in, const int* in_sizes, int n_in,
                              void* d_out, int out_size, void* d_ws, size_t ws_size,
                              hipStream_t stream) {
  const float* x   = (const float*)d_in[0];
  const float* pos = (const float*)d_in[1];
  const float* tq  = (const float*)d_in[2];
  // d_in[3] = batch (structure is fixed B x N contiguous, unused)
  const float* w1a = (const float*)d_in[4];  const float* b1a = (const float*)d_in[5];
  const float* w1b = (const float*)d_in[6];  const float* b1b = (const float*)d_in[7];
  const float* w2a = (const float*)d_in[8];  const float* b2a = (const float*)d_in[9];
  const float* w2b = (const float*)d_in[10]; const float* b2b = (const float*)d_in[11];
  const float* wl1 = (const float*)d_in[12]; const float* bl1 = (const float*)d_in[13];
  const float* wl2 = (const float*)d_in[14]; const float* bl2 = (const float*)d_in[15];
  const float* wm1 = (const float*)d_in[16]; const float* bm1 = (const float*)d_in[17];
  const float* wm2 = (const float*)d_in[18]; const float* bm2 = (const float*)d_in[19];

  char* ws = (char*)d_ws;
  size_t off = 0;
  auto alloc = [&](size_t bytes) -> void* {
    void* p = ws + off;
    off = (off + bytes + 255) & ~(size_t)255;
    return p;
  };
  const size_t R = (size_t)NB * NN;                       // 65536 rows
  float*          xx    = (float*)         alloc(R * 5 * 4);
  unsigned short* x1b   = (unsigned short*)alloc(R * 64 * 2);
  unsigned short* x2b   = (unsigned short*)alloc(R * 64 * 2);
  int*            idx1  = (int*)           alloc(R * NKNN * 4);
  int*            idx2  = (int*)           alloc(R * NKNN * 4);
  float*          sq    = (float*)         alloc(R * 4);
  unsigned short* comb  = (unsigned short*)alloc(R * 160 * 2);
  float*          gbuf  = (float*)         alloc((size_t)NB * 256 * 4);
  unsigned short* w1aT  = (unsigned short*)alloc(64  * 32  * 2);
  unsigned short* w1bT  = (unsigned short*)alloc(64  * 64  * 2);
  unsigned short* w2aT  = (unsigned short*)alloc(128 * 128 * 2);
  unsigned short* w2bT  = (unsigned short*)alloc(64  * 128 * 2);
  unsigned short* wl1T  = (unsigned short*)alloc(512 * 160 * 2);
  unsigned short* wl2T  = (unsigned short*)alloc(256 * 512 * 2);

  auto cvt = [&](const float* w, unsigned short* wt, int fin, int fout, int kpad) {
    int tot = fout * kpad;
    k_convw<<<(tot + 255) / 256, 256, 0, stream>>>(w, wt, fin, fout, kpad);
  };
  cvt(w1a, w1aT, 10, 64, 32);
  cvt(w1b, w1bT, 64, 64, 64);
  cvt(w2a, w2aT, 128, 128, 128);
  cvt(w2b, w2bT, 128, 64, 128);
  cvt(wl1, wl1T, 133, 512, 160);
  cvt(wl2, wl2T, 512, 256, 512);

  k_build_xx<<<(int)((R + 255) / 256), 256, 0, stream>>>(x, pos, tq, xx);
  k_knn1<<<NB, 512, 0, stream>>>(xx, idx1);
  k_edgeconv1<<<NB * 32, 256, 0, stream>>>(xx, idx1, w1aT, b1a, w1bT, b1b, x1b);
  k_sqnorm<<<(int)((R + 255) / 256), 256, 0, stream>>>(x1b, sq);
  k_knn2<<<NB * 16, 256, 0, stream>>>(x1b, sq, idx2);
  k_edgeconv2<<<NB * 64, 256, 0, stream>>>(x1b, idx2, w2aT, b2a, w2bT, b2b, x2b);
  k_comb<<<(int)((R * 160 + 255) / 256), 256, 0, stream>>>(xx, x1b, x2b, comb);
  k_initg<<<(NB * 256 + 255) / 256, 256, 0, stream>>>(gbuf);
  k_lin1<<<(int)(R / 32), 256, 0, stream>>>(comb, wl1T, bl1, wl2T, bl2, gbuf);
  k_head<<<NB, 128, 0, stream>>>(gbuf, wm1, bm1, wm2, bm2, (float*)d_out);
}
